// VRP_Decoder_3959959847657
// MI455X (gfx1250) — compile-verified, matrix operand order checked by reading, not run on hardware
//
#include <hip/hip_runtime.h>
#include <hip/hip_bf16.h>
#include <math.h>

// ---------------------------------------------------------------------------
// Problem constants (match reference)
// ---------------------------------------------------------------------------
#define BB   1024
#define PP   100
#define NN   101
#define EMB  128
#define NH   8
#define DKK  16

#define NPAD 112        // N padded to 7*16
#define PTIL 7          // ceil(100/16)
#define NTIL 7          // ceil(101/16)
#define KQ   160        // 132 padded to 5*32

typedef __attribute__((ext_vector_type(16))) _Float16 v16h;
typedef __attribute__((ext_vector_type(8)))  _Float16 v8h;
typedef __attribute__((ext_vector_type(4)))  _Float16 v4h;
typedef __attribute__((ext_vector_type(8)))  float    v8f;

// ---------------------------------------------------------------------------
// WMMA helpers (CDNA5 wave32, D = A(16x32 f16) * B(32x16 f16) + C(16x16 f32))
// ---------------------------------------------------------------------------
__device__ __forceinline__ v8f wmma_f16(v16h a, v16h b, v8f c) {
    return __builtin_amdgcn_wmma_f32_16x16x32_f16(
        /*neg_a=*/false, a, /*neg_b=*/false, b,
        /*c_mod=*/(short)0, c, /*reuse_a=*/false, /*reuse_b=*/false);
}

__device__ __forceinline__ v8f zero8() {
    v8f z = {0.f, 0.f, 0.f, 0.f, 0.f, 0.f, 0.f, 0.f};
    return z;
}

// A-operand (16x32, M x K) from row-major f16 [M][stride]; also valid for the
// B-operand (32x16, K x N) when B is stored "column-major by N", i.e. [N][stride]
// holding K along the row.  Per ISA 7.12.2: lanes 0-15 hold K {0..7,16..23},
// lanes 16-31 hold K {8..15,24..31}  ->  two aligned 16B LDS reads per lane.
__device__ __forceinline__ v16h frag_k32(const _Float16* base, int row0, int k0,
                                         int stride, int lane) {
    const int l  = lane & 15;
    const int hi = (lane >> 4) & 1;
    const _Float16* p = base + (size_t)(row0 + l) * stride + k0 + hi * 8;
    v8h lo = *(const v8h*)p;
    v8h hh = *(const v8h*)(p + 16);
    v16h r;
#pragma unroll
    for (int i = 0; i < 8; ++i) { r[i] = lo[i]; r[i + 8] = hh[i]; }
    return r;
}

// Same, but only K=0..15 are real (K=16..31 forced to zero) — used for the
// DK=16 QK^T contraction padded up to the 16x16x32 shape.
__device__ __forceinline__ v16h frag_k16(const _Float16* base, int row0, int k0,
                                         int stride, int lane) {
    const int l  = lane & 15;
    const int hi = (lane >> 4) & 1;
    const _Float16* p = base + (size_t)(row0 + l) * stride + k0 + hi * 8;
    v8h lo = *(const v8h*)p;
    v16h r;
#pragma unroll
    for (int i = 0; i < 8; ++i) { r[i] = lo[i]; r[i + 8] = (_Float16)0.f; }
    return r;
}

// ---------------------------------------------------------------------------
// Dynamic-LDS layout (see round-1 comments); total 229,888 B (< 320 KB WGP)
// ---------------------------------------------------------------------------
#define SMEM_PERSIST (28672 + 28672 + 32768 + 32768 + 512 + 40960)
#define SMEM_UNION   65536
#define SMEM_TOTAL   (SMEM_PERSIST + SMEM_UNION)

__global__ __launch_bounds__(256)
void vrp_decoder_kernel(const float* __restrict__ eln,    // [B,P,EMB]
                        const float* __restrict__ gload,  // [B,P]
                        const float* __restrict__ gtime,  // [B,P]
                        const float* __restrict__ glen,   // [B,P]
                        const float* __restrict__ gopen,  // [B,P]
                        const float* __restrict__ ninf,   // [B,P,N]
                        const float* __restrict__ enc,    // [B,N,EMB]
                        const float* __restrict__ Wq,     // [132,128]
                        const float* __restrict__ Wk,     // [128,128]
                        const float* __restrict__ Wv,     // [128,128]
                        const float* __restrict__ Wc,     // [128,128]
                        const float* __restrict__ bcv,    // [128]
                        float* __restrict__ out)          // [B,P,N]
{
    extern __shared__ char smem[];
    const int b    = blockIdx.x;
    const int tid  = threadIdx.x;
    const int wave = tid >> 5;     // 0..7
    const int lane = tid & 31;
    const int l    = lane & 15;
    const int hi   = (lane >> 4) & 1;

    _Float16* sEN  = (_Float16*)smem;            // [112][128]
    _Float16* sKT  = sEN  + NPAD * EMB;          // [112][128] K as [n][h*16+d]
    _Float16* sVT  = sKT  + NPAD * EMB;          // [128][128] V^T as [h*16+d][n]
    _Float16* sWcT = sVT  + EMB * EMB;           // [128][128]
    float*    sbc  = (float*)(sWcT + EMB * EMB); // [128]
    _Float16* sWqT = (_Float16*)(sbc + EMB);     // [128][160]
    char*     uni  = (char*)(sWqT + EMB * KQ);
    // phase 0/1 overlay
    _Float16* sWkT = (_Float16*)uni;             // [128][128]
    _Float16* sWvT = sWkT + EMB * EMB;           // [128][128]
    // P-loop overlay
    _Float16* sA   = (_Float16*)uni;             // [16][160]  input_cat tile
    _Float16* sQ   = sA   + 16 * KQ;             // [16][128]
    _Float16* sWgt = sQ   + 16 * EMB;            // [8][16][128] exp weights
    _Float16* sOC  = sWgt + NH * 16 * EMB;       // [16][128]  out_concat
    _Float16* sMH  = sOC  + 16 * EMB;            // [16][128]  mh
    float*    sS2  = (float*)(sMH + 16 * EMB);   // [16][112]  score2
    float*    sMask= sS2  + 16 * NPAD;           // [16][112]
    float*    sRed = sMask + 16 * NPAD;          // [16][2]

    // ---------------- phase 0: stage encoded_nodes + Wk/Wv -----------------
    // encoded_nodes: 512B global_load_b128 per instr, contiguous b64 LDS store
    for (int i4 = tid; i4 < NPAD * (EMB / 4); i4 += 256) {
        int r = i4 >> 5, c4 = (i4 & 31) * 4;
        float4 v = make_float4(0.f, 0.f, 0.f, 0.f);
        if (r < NN) v = *(const float4*)&enc[((size_t)b * NN + r) * EMB + c4];
        v4h h; h[0] = (_Float16)v.x; h[1] = (_Float16)v.y;
        h[2] = (_Float16)v.z; h[3] = (_Float16)v.w;
        *(v4h*)&sEN[r * EMB + c4] = h;
    }
    // Wk/Wv: b128 loads along the out-dim, transposed scatter into LDS
    for (int i4 = tid; i4 < EMB * (EMB / 4); i4 += 256) {
        int k = i4 >> 5, n4 = (i4 & 31) * 4;
        float4 wk = *(const float4*)&Wk[k * EMB + n4];
        float4 wv = *(const float4*)&Wv[k * EMB + n4];
        sWkT[(n4 + 0) * EMB + k] = (_Float16)wk.x;
        sWkT[(n4 + 1) * EMB + k] = (_Float16)wk.y;
        sWkT[(n4 + 2) * EMB + k] = (_Float16)wk.z;
        sWkT[(n4 + 3) * EMB + k] = (_Float16)wk.w;
        sWvT[(n4 + 0) * EMB + k] = (_Float16)wv.x;
        sWvT[(n4 + 1) * EMB + k] = (_Float16)wv.y;
        sWvT[(n4 + 2) * EMB + k] = (_Float16)wv.z;
        sWvT[(n4 + 3) * EMB + k] = (_Float16)wv.w;
    }
    for (int i = tid; i < EMB * 16; i += 256) { // zero V^T cols n=112..127
        int r = i >> 4, c = i & 15;
        sVT[r * EMB + NPAD + c] = (_Float16)0.f;
    }
    __syncthreads();

    // ---------------- phase 1: K = EN@Wk, V = EN@Wv ------------------------
    for (int t = wave; t < 2 * NTIL * 8; t += 8) {
        const bool isK = (t < NTIL * 8);
        const int  tt  = isK ? t : t - NTIL * 8;
        const int  mt  = tt >> 3;               // n-tile 0..6
        const int  ct  = tt & 7;                // col-tile 0..7 (h*16+d)
        const _Float16* wT = isK ? sWkT : sWvT;
        v8f acc = zero8();
#pragma unroll
        for (int kt = 0; kt < 4; ++kt) {
            v16h a  = frag_k32(sEN, mt * 16, kt * 32, EMB, lane);
            v16h bf = frag_k32(wT, ct * 16, kt * 32, EMB, lane);
            acc = wmma_f16(a, bf, acc);
        }
        if (isK) {                               // sKT[n][col] (strided)
#pragma unroll
            for (int r = 0; r < 8; ++r)
                sKT[(mt * 16 + r + hi * 8) * EMB + ct * 16 + l] = (_Float16)acc[r];
        } else {                                 // sVT[col][n]: 8 contiguous
            v8h h;                               //   halves -> one b128 store
#pragma unroll
            for (int r = 0; r < 8; ++r) h[r] = (_Float16)acc[r];
            *(v8h*)&sVT[(ct * 16 + l) * EMB + mt * 16 + hi * 8] = h;
        }
    }
    __syncthreads();

    // ---------------- phase 2: stage Wq^T (K-pad to 160), Wc^T, bc ---------
    for (int i4 = tid; i4 < (EMB + 4) * (EMB / 4); i4 += 256) {
        int k = i4 >> 5, n4 = (i4 & 31) * 4;     // k in 0..131
        float4 wq = *(const float4*)&Wq[k * EMB + n4];
        sWqT[(n4 + 0) * KQ + k] = (_Float16)wq.x;
        sWqT[(n4 + 1) * KQ + k] = (_Float16)wq.y;
        sWqT[(n4 + 2) * KQ + k] = (_Float16)wq.z;
        sWqT[(n4 + 3) * KQ + k] = (_Float16)wq.w;
    }
    for (int i = tid; i < EMB * (KQ - (EMB + 4)); i += 256) {
        int n = i / (KQ - (EMB + 4)), k = (EMB + 4) + i % (KQ - (EMB + 4));
        sWqT[n * KQ + k] = (_Float16)0.f;        // zero K-pad 132..159
    }
    for (int i4 = tid; i4 < EMB * (EMB / 4); i4 += 256) {
        int k = i4 >> 5, n4 = (i4 & 31) * 4;
        float4 wc = *(const float4*)&Wc[k * EMB + n4];
        sWcT[(n4 + 0) * EMB + k] = (_Float16)wc.x;
        sWcT[(n4 + 1) * EMB + k] = (_Float16)wc.y;
        sWcT[(n4 + 2) * EMB + k] = (_Float16)wc.z;
        sWcT[(n4 + 3) * EMB + k] = (_Float16)wc.w;
    }
    if (tid < EMB) sbc[tid] = bcv[tid];
    for (int i = tid; i < NH * 16 * EMB; i += 256) sWgt[i] = (_Float16)0.f;
    __syncthreads();

    // ---------------- P-tile loop ------------------------------------------
    for (int pt = 0; pt < PTIL; ++pt) {
        const int p0 = pt * 16;

        // input_cat tile: embedding part via b128 loads
        for (int i4 = tid; i4 < 16 * (EMB / 4); i4 += 256) {
            int r = i4 >> 5, c4 = (i4 & 31) * 4;
            int p = p0 + r;
            float4 v = make_float4(0.f, 0.f, 0.f, 0.f);
            if (p < PP) v = *(const float4*)&eln[((size_t)b * PP + p) * EMB + c4];
            v4h h; h[0] = (_Float16)v.x; h[1] = (_Float16)v.y;
            h[2] = (_Float16)v.z; h[3] = (_Float16)v.w;
            *(v4h*)&sA[r * KQ + c4] = h;
        }
        // the 4 scalar features + K-pad (cols 128..159)
        for (int i = tid; i < 16 * 32; i += 256) {
            int r = i >> 5, c = EMB + (i & 31);
            int p = p0 + r;
            float v = 0.f;
            if (p < PP) {
                if      (c == EMB)     v = gload[(size_t)b * PP + p];
                else if (c == EMB + 1) v = gtime[(size_t)b * PP + p];
                else if (c == EMB + 2) v = glen [(size_t)b * PP + p];
                else if (c == EMB + 3) v = gopen[(size_t)b * PP + p];
            }
            sA[r * KQ + c] = (_Float16)v;
        }
        // mask tile
        for (int i = tid; i < 16 * NPAD; i += 256) {
            int r = i / NPAD, n = i % NPAD;
            int p = p0 + r;
            float v;
            if (n >= NN)      v = -1e30f;                       // N padding
            else if (p < PP)  v = ninf[((size_t)b * PP + p) * NN + n];
            else              v = 0.f;                          // dead rows
            sMask[i] = v;
        }
        __syncthreads();

        // ---- Q tile: wave w computes output cols w*16..w*16+15, K = 160 ---
        {
            v8f acc = zero8();
#pragma unroll
            for (int kt = 0; kt < 5; ++kt) {
                v16h a  = frag_k32(sA, 0, kt * 32, KQ, lane);
                v16h bf = frag_k32(sWqT, wave * 16, kt * 32, KQ, lane);
                acc = wmma_f16(a, bf, acc);
            }
#pragma unroll
            for (int r = 0; r < 8; ++r)
                sQ[(r + hi * 8) * EMB + wave * 16 + l] = (_Float16)acc[r];
        }
        __syncthreads();

        // ---- attention: wave == head --------------------------------------
        float rinv[8];
        {
            const int h = wave;
            v16h aq = frag_k16(sQ, 0, h * 16, EMB, lane);
            v8f sc[NTIL];
#pragma unroll
            for (int nt = 0; nt < NTIL; ++nt) {
                v16h bk = frag_k16(sKT, nt * 16, h * 16, EMB, lane);
                sc[nt] = wmma_f16(aq, bk, zero8());
            }
            // scale 1/sqrt(16) + mask   (D elem (r,lane) = row r+hi*8, col l)
#pragma unroll
            for (int nt = 0; nt < NTIL; ++nt)
#pragma unroll
                for (int r = 0; r < 8; ++r)
                    sc[nt][r] = sc[nt][r] * 0.25f +
                                sMask[(r + hi * 8) * NPAD + nt * 16 + l];
            // row-max across the 7 tiles, then across 16 lanes of the half
            float m[8];
#pragma unroll
            for (int r = 0; r < 8; ++r) {
                m[r] = sc[0][r];
#pragma unroll
                for (int nt = 1; nt < NTIL; ++nt) m[r] = fmaxf(m[r], sc[nt][r]);
            }
#pragma unroll
            for (int off = 1; off < 16; off <<= 1)
#pragma unroll
                for (int r = 0; r < 8; ++r)
                    m[r] = fmaxf(m[r], __shfl_xor(m[r], off, 32));
            // exp (unnormalized) -> LDS as the AV A-operand; row sums in regs
            float s[8];
#pragma unroll
            for (int r = 0; r < 8; ++r) s[r] = 0.f;
#pragma unroll
            for (int nt = 0; nt < NTIL; ++nt)
#pragma unroll
                for (int r = 0; r < 8; ++r) {
                    float e = __expf(sc[nt][r] - m[r]);
                    s[r] += e;
                    sWgt[(h * 16 + r + hi * 8) * EMB + nt * 16 + l] = (_Float16)e;
                }
#pragma unroll
            for (int off = 1; off < 16; off <<= 1)
#pragma unroll
                for (int r = 0; r < 8; ++r) s[r] += __shfl_xor(s[r], off, 32);
#pragma unroll
            for (int r = 0; r < 8; ++r) rinv[r] = 1.f / s[r];
        }
        __syncthreads();   // sWgt visible

        // ---- AV + fold 1/rowsum -> out_concat -----------------------------
        {
            const int h = wave;
            v8f o = zero8();
#pragma unroll
            for (int kt = 0; kt < 4; ++kt) {
                v16h aw = frag_k32(sWgt + h * 16 * EMB, 0, kt * 32, EMB, lane);
                v16h bv = frag_k32(sVT, h * 16, kt * 32, EMB, lane);
                o = wmma_f16(aw, bv, o);
            }
#pragma unroll
            for (int r = 0; r < 8; ++r)
                sOC[(r + hi * 8) * EMB + h * 16 + l] = (_Float16)(o[r] * rinv[r]);
        }
        __syncthreads();

        // ---- mh = out_concat @ Wc + bc : wave w -> emb cols w*16.. --------
        {
            v8f acc = zero8();
#pragma unroll
            for (int kt = 0; kt < 4; ++kt) {
                v16h a  = frag_k32(sOC, 0, kt * 32, EMB, lane);
                v16h bf = frag_k32(sWcT, wave * 16, kt * 32, EMB, lane);
                acc = wmma_f16(a, bf, acc);
            }
            float bias = sbc[wave * 16 + l];
#pragma unroll
            for (int r = 0; r < 8; ++r)
                sMH[(r + hi * 8) * EMB + wave * 16 + l] = (_Float16)(acc[r] + bias);
        }
        __syncthreads();

        // ---- score2 = mh @ EN^T / sqrt(EMB); 10*tanh + mask ---------------
        if (wave < NTIL) {
            v8f acc = zero8();
#pragma unroll
            for (int kt = 0; kt < 4; ++kt) {
                v16h a  = frag_k32(sMH, 0, kt * 32, EMB, lane);
                v16h bf = frag_k32(sEN, wave * 16, kt * 32, EMB, lane);
                acc = wmma_f16(a, bf, acc);
            }
#pragma unroll
            for (int r = 0; r < 8; ++r) {
                float v = acc[r] * (1.0f / 11.313708498984761f);
                v = 10.0f * tanhf(v) +
                    sMask[(r + hi * 8) * NPAD + wave * 16 + l];
                sS2[(r + hi * 8) * NPAD + wave * 16 + l] = v;
            }
        }
        __syncthreads();

        // ---- final softmax over N (2 lanes per row), write probs ----------
        if (wave == 0) {
            // lane l handles row l, half hi: columns [hi*56, hi*56+56)
            float m = -3.4e38f;
            for (int n = hi * 56; n < hi * 56 + 56; ++n)
                m = fmaxf(m, sS2[l * NPAD + n]);
            m = fmaxf(m, __shfl_xor(m, 16, 32));
            float sum = 0.f;
            for (int n = hi * 56; n < hi * 56 + 56; ++n)
                sum += __expf(sS2[l * NPAD + n] - m);
            sum += __shfl_xor(sum, 16, 32);
            if (hi == 0) {
                sRed[l * 2 + 0] = m;
                sRed[l * 2 + 1] = 1.f / sum;
            }
        }
        __syncthreads();
        for (int i = tid; i < 16 * NN; i += 256) {
            int r = i / NN, n = i % NN;
            int p = p0 + r;
            if (p < PP)
                out[((size_t)b * PP + p) * NN + n] =
                    __expf(sS2[r * NPAD + n] - sRed[r * 2]) * sRed[r * 2 + 1];
        }
        __syncthreads();   // protect sA/sMask/sS2 before next P tile
    }
}

// ---------------------------------------------------------------------------
extern "C" void kernel_launch(void* const* d_in, const int* in_sizes, int n_in,
                              void* d_out, int out_size, void* d_ws, size_t ws_size,
                              hipStream_t stream) {
    (void)in_sizes; (void)n_in; (void)out_size; (void)d_ws; (void)ws_size;
    const float* eln   = (const float*)d_in[0];
    const float* gload = (const float*)d_in[1];
    const float* gtime = (const float*)d_in[2];
    const float* glen  = (const float*)d_in[3];
    const float* gopen = (const float*)d_in[4];
    const float* ninf  = (const float*)d_in[5];
    const float* enc   = (const float*)d_in[6];
    const float* Wq    = (const float*)d_in[7];
    const float* Wk    = (const float*)d_in[8];
    const float* Wv    = (const float*)d_in[9];
    const float* Wc    = (const float*)d_in[10];
    const float* bcv   = (const float*)d_in[11];
    float* out = (float*)d_out;

    hipFuncSetAttribute((const void*)vrp_decoder_kernel,
                        hipFuncAttributeMaxDynamicSharedMemorySize,
                        SMEM_TOTAL);
    vrp_decoder_kernel<<<dim3(BB), dim3(256), SMEM_TOTAL, stream>>>(
        eln, gload, gtime, glen, gopen, ninf, enc, Wq, Wk, Wv, Wc, bcv, out);
}
